// CoreAttention_31559419691189
// MI455X (gfx1250) — compile-verified, matrix operand order checked by reading, not run on hardware
//
#include <hip/hip_runtime.h>

typedef __attribute__((ext_vector_type(16))) __bf16 v16bf;
typedef __attribute__((ext_vector_type(8)))  __bf16 v8bf;
typedef __attribute__((ext_vector_type(8)))  float  v8f;
typedef __attribute__((ext_vector_type(4)))  unsigned int u32x4;
typedef __attribute__((ext_vector_type(8)))  int i32x8;
typedef __attribute__((ext_vector_type(4)))  int i32x4;

#define WMMA_BF16(A, B, C) \
  __builtin_amdgcn_wmma_f32_16x16x32_bf16(false, (A), false, (B), (short)0, (C), false, false)

#define CAT16(lo, hi) \
  __builtin_shufflevector((lo), (hi), 0, 1, 2, 3, 4, 5, 6, 7, 8, 9, 10, 11, 12, 13, 14, 15)

namespace {
constexpr int   kS           = 2048;
constexpr int   kD           = 128;
constexpr int   kQPerBlock   = 128;            // 8 waves x 16 query rows
constexpr int   kWavesPerBlk = 8;
constexpr int   kLdsStride   = 132;  // fp32 TDM rows: 128 + 4 dword pad
constexpr int   kKbfStride   = 136;  // bf16 K rows: 272B (16B aligned, 4-bank rotate)
constexpr int   kVtStride    = 40;   // bf16 V^T rows: 80B (16B aligned, 20-bank rotate)
// (1/sqrt(128)) * log2(e): fold softmax scale + exp2 conversion into Q at load
constexpr float kQScale      = 0.088388347648318447f * 1.4426950408889634f;
constexpr float kNegBig      = -1.0e30f;
}

// Issue one TDM 2D tile load: 32 rows x 128 fp32 cols from a row-major
// [S, 128] tensor into LDS, padding +4 dwords per row (row stride = 132 dw).
__device__ __forceinline__ void tdm_load_tile32x128(unsigned ldsByteOff,
                                                    const float* gtile) {
  const unsigned long long ga = (unsigned long long)(uintptr_t)gtile;
  u32x4 g0;
  g0[0] = 1u;                                        // count=1 user descriptor
  g0[1] = ldsByteOff;                                // lds_addr
  g0[2] = (unsigned)(ga & 0xFFFFFFFFu);              // global_addr[31:0]
  g0[3] = (unsigned)((ga >> 32) & 0x01FFFFFFu)       // global_addr[56:32]
        | (2u << 30);                                // type = 2 ("image")
  i32x8 g1;
  g1[0] = (2 << 16)      // data_size = 4 bytes
        | (1 << 20)      // pad_enable
        | (6 << 22)      // pad_interval: every 128 dwords
        | (3 << 25);     // pad_amount: 4 dwords
  g1[1] = (int)(128u << 16);    // tensor_dim0 = 128 (d-columns)
  g1[2] = (int)(2048u << 16);   // tensor_dim1 = 2048 (rows)
  g1[3] = (int)(128u << 16);    // tile_dim0 = 128
  g1[4] = 32;                   // tile_dim1 = 32 rows
  g1[5] = 128;                  // tensor_dim0_stride = 128
  g1[6] = 0;
  g1[7] = 0;
  i32x4 z4;
  z4[0] = 0; z4[1] = 0; z4[2] = 0; z4[3] = 0;
  i32x8 z8;
  #pragma unroll
  for (int i = 0; i < 8; ++i) z8[i] = 0;
#if __has_builtin(__builtin_amdgcn_tensor_load_to_lds)
  __builtin_amdgcn_tensor_load_to_lds(g0, g1, z4, z4, z8, 0);
#endif
}

__global__ __launch_bounds__(256)
void fa_causal_tdm_31559419691189(const float* __restrict__ Q,
                                  const float* __restrict__ K,
                                  const float* __restrict__ V,
                                  float* __restrict__ O)
{
  __shared__ float  Kf[32 * kLdsStride];             // TDM target (fp32)  16.5 KB
  __shared__ float  Vf[32 * kLdsStride];             // TDM target (fp32)  16.5 KB
  __shared__ __bf16 Kbf[32 * kKbfStride];            // K chunk, bf16       8.5 KB
  __shared__ __bf16 VT[128 * kVtStride];             // V chunk, bf16, T   10.0 KB
  __shared__ __bf16 ldsP[kWavesPerBlk][16 * 32];     // per-wave P tiles    8.0 KB

  const int tid  = threadIdx.x;
  const int lane = tid & 31;
  const int wave = tid >> 5;
  const int hi   = lane >> 4;   // which 16-lane half of the wave
  const int ln   = lane & 15;

  const int bh  = blockIdx.x >> 4;            // B*H = 64
  const int qb  = blockIdx.x & 15;            // 16 query blocks of 128
  const int q0w = qb * kQPerBlock + wave * 16;

  const float* __restrict__ Qb = Q + (long long)bh * kS * kD;
  const float* __restrict__ Kb = K + (long long)bh * kS * kD;
  const float* __restrict__ Vb = V + (long long)bh * kS * kD;

  // ---- Q tile -> 4 A-fragments (16x32 bf16), pre-scaled ----
  // A layout (16-bit 16x32): lanes 0-15: row=ln, K={0..7,16..23};
  //                          lanes 16-31: row=ln, K={8..15,24..31}
  v16bf aQ[4];
  {
    const float* qrow = Qb + (long long)(q0w + ln) * kD;
    #pragma unroll
    for (int kb = 0; kb < 4; ++kb) {
      const int c0 = kb * 32 + hi * 8;
      #pragma unroll
      for (int e = 0; e < 8; ++e) {
        aQ[kb][e]     = (__bf16)(qrow[c0 + e]      * kQScale);
        aQ[kb][e + 8] = (__bf16)(qrow[c0 + 16 + e] * kQScale);
      }
    }
  }

  v8f acc[8];
  #pragma unroll
  for (int t = 0; t < 8; ++t)
    #pragma unroll
    for (int r = 0; r < 8; ++r) acc[t][r] = 0.0f;

  float mrow[8], lrow[8];
  #pragma unroll
  for (int r = 0; r < 8; ++r) { mrow[r] = kNegBig; lrow[r] = 0.0f; }

  const unsigned kfOff = (unsigned)(uintptr_t)&Kf[0];
  const unsigned vfOff = (unsigned)(uintptr_t)&Vf[0];
  const int kendBlk = qb * kQPerBlock + kQPerBlock;  // block-uniform trip count

#if __has_builtin(__builtin_amdgcn_tensor_load_to_lds)
  if (wave == 0) {   // prologue: DMA first chunk
    tdm_load_tile32x128(kfOff, Kb);
    tdm_load_tile32x128(vfOff, Vb);
  }
#endif

  for (int kc = 0; kc < kendBlk; kc += 32) {
#if __has_builtin(__builtin_amdgcn_tensor_load_to_lds)
    if (wave == 0) __builtin_amdgcn_s_wait_tensorcnt(0);
#else
    for (int i = tid * 4; i < 32 * kD; i += 256 * 4) {   // fallback staging
      const int row = i >> 7, col = i & 127;
      #pragma unroll
      for (int u = 0; u < 4; ++u) {
        Kf[row * kLdsStride + col + u] = Kb[(long long)(kc + row) * kD + col + u];
        Vf[row * kLdsStride + col + u] = Vb[(long long)(kc + row) * kD + col + u];
      }
    }
#endif
    __syncthreads();   // fp32 chunk staged & visible

    // ---- cooperative one-shot fp32 -> bf16 conversion ----
    {
      // K: natural [key][dcol], 16 elems per thread
      const int krow = tid >> 3;
      const int kcg  = (tid & 7) * 16;
      const float* ksrc = &Kf[krow * kLdsStride + kcg];
      __bf16*      kdst = &Kbf[krow * kKbfStride + kcg];
      #pragma unroll
      for (int h = 0; h < 2; ++h) {
        v8bf p;
        #pragma unroll
        for (int u = 0; u < 8; ++u) p[u] = (__bf16)ksrc[h * 8 + u];
        *(v8bf*)(kdst + h * 8) = p;
      }
      // V: transposed [dcol][key], 16 elems per thread
      const int dcol = tid >> 1;
      const int kh   = (tid & 1) * 16;
      __bf16* vdst = &VT[dcol * kVtStride + kh];
      #pragma unroll
      for (int h = 0; h < 2; ++h) {
        v8bf p;
        #pragma unroll
        for (int u = 0; u < 8; ++u)
          p[u] = (__bf16)Vf[(kh + h * 8 + u) * kLdsStride + dcol];
        *(v8bf*)(vdst + h * 8) = p;
      }
    }
    __syncthreads();   // bf16 tiles ready; fp32 buffers now dead

#if __has_builtin(__builtin_amdgcn_tensor_load_to_lds)
    if (wave == 0 && kc + 32 < kendBlk) {   // overlap next DMA with compute
      tdm_load_tile32x128(kfOff, Kb + (long long)(kc + 32) * kD);
      tdm_load_tile32x128(vfOff, Vb + (long long)(kc + 32) * kD);
    }
#endif

    if (kc < q0w + 16) {   // wave participates (causal range)
      // ---- scores: two 16x16 tiles, chained over K-dim = 128 ----
      v8f s0, s1;
      #pragma unroll
      for (int r = 0; r < 8; ++r) { s0[r] = 0.0f; s1[r] = 0.0f; }

      const __bf16* kr0 = &Kbf[ln * kKbfStride];          // keys kc..kc+15
      const __bf16* kr1 = &Kbf[(16 + ln) * kKbfStride];   // keys kc+16..kc+31
      #pragma unroll
      for (int kb = 0; kb < 4; ++kb) {
        const int c0 = kb * 32 + hi * 8;
        const v16bf b0 = CAT16(*(const v8bf*)(kr0 + c0), *(const v8bf*)(kr0 + c0 + 16));
        const v16bf b1 = CAT16(*(const v8bf*)(kr1 + c0), *(const v8bf*)(kr1 + c0 + 16));
        s0 = WMMA_BF16(aQ[kb], b0, s0);
        s1 = WMMA_BF16(aQ[kb], b1, s1);
      }

      // ---- causal mask (only diagonal-straddling chunks) ----
      if (kc + 31 > q0w) {
        #pragma unroll
        for (int r = 0; r < 8; ++r) {
          const int q = q0w + r + hi * 8;
          if (kc + ln > q)      s0[r] = kNegBig;
          if (kc + 16 + ln > q) s1[r] = kNegBig;
        }
      }

      // ---- online softmax update (reductions across 16-lane halves) ----
      v8f   p0, p1;
      float corr[8];
      #pragma unroll
      for (int r = 0; r < 8; ++r) {
        float mx = fmaxf(s0[r], s1[r]);
        #pragma unroll
        for (int off = 8; off > 0; off >>= 1)
          mx = fmaxf(mx, __shfl_xor(mx, off, 32));
        const float mnew = fmaxf(mrow[r], mx);
        const float c    = exp2f(mrow[r] - mnew);
        const float e0   = exp2f(s0[r] - mnew);
        const float e1   = exp2f(s1[r] - mnew);
        float rs = e0 + e1;
        #pragma unroll
        for (int off = 8; off > 0; off >>= 1)
          rs += __shfl_xor(rs, off, 32);
        mrow[r] = mnew;
        lrow[r] = lrow[r] * c + rs;
        corr[r] = c;
        p0[r] = e0;
        p1[r] = e1;
      }
      #pragma unroll
      for (int t = 0; t < 8; ++t)
        #pragma unroll
        for (int r = 0; r < 8; ++r) acc[t][r] *= corr[r];

      // ---- re-layout P (C-layout) -> A-layout via per-wave LDS tile ----
      __bf16* lp = &ldsP[wave][0];
      #pragma unroll
      for (int r = 0; r < 8; ++r) {
        const int row = r + hi * 8;
        lp[row * 32 + ln]      = (__bf16)p0[r];
        lp[row * 32 + 16 + ln] = (__bf16)p1[r];
      }
      asm volatile("s_wait_dscnt 0" ::: "memory");  // per-wave RAW on LDS
      const __bf16* pb = lp + ln * 32 + hi * 8;
      const v16bf aP = CAT16(*(const v8bf*)(pb), *(const v8bf*)(pb + 16));

      // ---- PV: accumulate into 8 resident C-tiles ----
      // B layout (32x16): lanes 0-15: col=ln, K={0..7,16..23};
      //                   lanes 16-31: col=ln, K={8..15,24..31}
      #pragma unroll
      for (int t = 0; t < 8; ++t) {
        const __bf16* vt = &VT[(t * 16 + ln) * kVtStride + hi * 8];
        const v16bf bV = CAT16(*(const v8bf*)(vt), *(const v8bf*)(vt + 16));
        acc[t] = WMMA_BF16(aP, bV, acc[t]);
      }
    }

    __syncthreads();   // all waves done with bf16 tiles before next convert
  }

  // ---- epilogue: normalize and store fp32 output ----
  float inv[8];
  #pragma unroll
  for (int r = 0; r < 8; ++r) inv[r] = 1.0f / lrow[r];

  float* Ob = O + (long long)bh * kS * kD;
  #pragma unroll
  for (int r = 0; r < 8; ++r) {
    float* orow = Ob + (long long)(q0w + r + hi * 8) * kD + ln;
    #pragma unroll
    for (int t = 0; t < 8; ++t)
      orow[t * 16] = acc[t][r] * inv[r];
  }
}

extern "C" void kernel_launch(void* const* d_in, const int* in_sizes, int n_in,
                              void* d_out, int out_size, void* d_ws, size_t ws_size,
                              hipStream_t stream) {
  const float* Q = (const float*)d_in[0];
  const float* K = (const float*)d_in[1];
  const float* V = (const float*)d_in[2];
  // d_in[3] is the causal bool mask; causality is hardcoded in-kernel.
  float* O = (float*)d_out;

  const int B = 4, H = 16;
  dim3 grid(B * H * (kS / kQPerBlock));   // 64 * 16 = 1024 blocks
  dim3 block(kWavesPerBlk * 32);          // 256 threads = 8 wave32

  hipLaunchKernelGGL(fa_causal_tdm_31559419691189, grid, block, 0, stream,
                     Q, K, V, O);
}
